// CrossFormerLayer_62646392980065
// MI455X (gfx1250) — matrix operand order repaired and checked by reading, not compiled, not run on hardware
//
#include <hip/hip_runtime.h>
#include <math.h>

// ---------------------------------------------------------------------------
// CrossFormer layer for MI455X (gfx1250, wave32).
//  - All GEMMs: bf16 operands, fp32 accum, V_WMMA_F32_16X16X32_BF16.
//  - TN layout everywhere (B stored [N][K]) -> contiguous 16B rows, so tiles
//    are staged with GLOBAL_LOAD_ASYNC_TO_LDS_B128 (ASYNCcnt) on the hot path.
//  - Double-buffered LDS, one barrier + one s_wait_asynccnt per K-step (64).
// ---------------------------------------------------------------------------

typedef float    v8f   __attribute__((ext_vector_type(8)));
typedef __bf16   v16bf __attribute__((ext_vector_type(16)));
typedef unsigned uvec4 __attribute__((ext_vector_type(4)));

union FragBF { uvec4 u[2]; v16bf v; };

__device__ __forceinline__ unsigned short f2bf(float f) {
  union { float f; unsigned u; } v; v.f = f;
  unsigned r = v.u + 0x7FFFu + ((v.u >> 16) & 1u);   // round-to-nearest-even
  return (unsigned short)(r >> 16);
}
__device__ __forceinline__ float bf2f(unsigned short h) {
  union { unsigned u; float f; } v; v.u = ((unsigned)h) << 16;
  return v.f;
}

// CDNA5 async DMA: copy 16B/lane from global to LDS, tracked by ASYNCcnt.
__device__ __forceinline__ void async_b128(unsigned lds_byte_off,
                                           const unsigned short* g) {
  asm volatile("global_load_async_to_lds_b128 %0, %1, off"
               :: "v"(lds_byte_off), "v"(g) : "memory");
}
__device__ __forceinline__ void wait_async0() {
  asm volatile("s_wait_asynccnt 0" ::: "memory");
}

// ---------------------------------------------------------------------------
// Batched TN GEMM:  C[z] = epilogue( alpha * A[z] @ B[z]^T )
//   A: bf16 [M x K] row-major (lda);  B: bf16 [N x K] row-major (ldb)
//   C: fp32 or bf16; OUTTRANS stores element (m,n) at C[n*ldc + m].
//   z = b*Hnum + h; element strides sXb/sXh per batch/head.
//   Block tile 128x64x64, 8 waves (4x2), wave tile 32x32, 8 WMMA / K-step.
//   GUARD=false: all tiles full -> async global->LDS staging.
// ---------------------------------------------------------------------------
#define BM 128
#define BN 64
#define BK 64

template <bool GUARD, bool OUTBF16, bool OUTTRANS>
__global__ __launch_bounds__(256) void gemm_tn_wmma(
    int M, int N, int K,
    const unsigned short* __restrict__ A, int lda, long long sAb, long long sAh,
    const unsigned short* __restrict__ Bm, int ldb, long long sBb, long long sBh,
    void* __restrict__ Cp, int ldc, long long sCb, long long sCh,
    const float* __restrict__ bias,
    const float* __restrict__ resid, int ldr,
    float alpha, int relu, int Hnum) {
  __shared__ __align__(16) unsigned short As[2][BM * BK];
  __shared__ __align__(16) unsigned short Bs[2][BN * BK];

  const int z  = blockIdx.z;
  const int bb = z / Hnum;
  const int hh = z - bb * Hnum;
  A  += (size_t)bb * sAb + (size_t)hh * sAh;
  Bm += (size_t)bb * sBb + (size_t)hh * sBh;
  const size_t coff = (size_t)bb * sCb + (size_t)hh * sCh;
  float*          Cf = (float*)Cp + coff;
  unsigned short* Cb = (unsigned short*)Cp + coff;

  const int tid  = threadIdx.x;
  const int lane = tid & 31;
  const int wid  = tid >> 5;
  const int wm   = (wid & 3) << 5;   // wave row offset in block tile
  const int wn   = (wid >> 2) << 5;  // wave col offset
  const int m0   = blockIdx.y * BM;
  const int n0   = blockIdx.x * BN;
  const int lrow = lane & 15;
  const bool hi  = lane >= 16;

  v8f acc[2][2];
#pragma unroll
  for (int i = 0; i < 2; i++)
#pragma unroll
    for (int j = 0; j < 2; j++)
#pragma unroll
      for (int e = 0; e < 8; e++) acc[i][j][e] = 0.0f;

  const uvec4 zero4 = {0u, 0u, 0u, 0u};
  const int nk = K / BK;

  // LDS byte bases for async staging
  unsigned asb[2], bsb[2];
  asb[0] = (unsigned)(unsigned long long)(&As[0][0]);
  asb[1] = (unsigned)(unsigned long long)(&As[1][0]);
  bsb[0] = (unsigned)(unsigned long long)(&Bs[0][0]);
  bsb[1] = (unsigned)(unsigned long long)(&Bs[1][0]);

  // -------- staging helpers ------------------------------------------------
  auto stage_async = [&](int st, int k0) {
#pragma unroll
    for (int p = 0; p < 4; p++) {                 // A: 128x64 = 4 passes
      int idx = tid * 8 + p * 2048;
      int r = idx >> 6, kk = idx & 63;
      async_b128(asb[st] + idx * 2, A + (size_t)(m0 + r) * lda + k0 + kk);
    }
#pragma unroll
    for (int p = 0; p < 2; p++) {                 // B: 64x64 = 2 passes
      int idx = tid * 8 + p * 2048;
      int n = idx >> 6, kk = idx & 63;
      async_b128(bsb[st] + idx * 2, Bm + (size_t)(n0 + n) * ldb + k0 + kk);
    }
  };
  uvec4 pa[4], pb[2];
  auto fetch_guarded = [&](int k0) {
#pragma unroll
    for (int p = 0; p < 4; p++) {
      int idx = tid * 8 + p * 2048;
      int r = idx >> 6, kk = idx & 63;
      pa[p] = zero4;
      if (m0 + r < M)
        pa[p] = *(const uvec4*)(A + (size_t)(m0 + r) * lda + k0 + kk);
    }
#pragma unroll
    for (int p = 0; p < 2; p++) {
      int idx = tid * 8 + p * 2048;
      int n = idx >> 6, kk = idx & 63;
      pb[p] = zero4;
      if (n0 + n < N)
        pb[p] = *(const uvec4*)(Bm + (size_t)(n0 + n) * ldb + k0 + kk);
    }
  };
  auto commit_guarded = [&](int st) {
#pragma unroll
    for (int p = 0; p < 4; p++) *(uvec4*)(&As[st][tid * 8 + p * 2048]) = pa[p];
#pragma unroll
    for (int p = 0; p < 2; p++) *(uvec4*)(&Bs[st][tid * 8 + p * 2048]) = pb[p];
  };

  // -------- prologue: stage 0 ---------------------------------------------
  if (GUARD) {
    fetch_guarded(0);
    commit_guarded(0);
  } else {
    stage_async(0, 0);
    wait_async0();
  }
  __syncthreads();

  // -------- main loop ------------------------------------------------------
  for (int kt = 0; kt < nk; ++kt) {
    const int cur = kt & 1;
    const bool more = (kt + 1 < nk);
    if (more) {
      if (GUARD) fetch_guarded((kt + 1) * BK);      // loads in flight
      else       stage_async(cur ^ 1, (kt + 1) * BK);
    }
    // compute from stage cur: 2 K-slices of 32 -> 8 WMMA
#pragma unroll
    for (int ks = 0; ks < BK; ks += 32) {
      FragBF a[2], bfr[2];
#pragma unroll
      for (int i = 0; i < 2; i++) {
        const unsigned short* s = &As[cur][(wm + i * 16 + lrow) * BK + ks];
        a[i].u[0] = *(const uvec4*)(s + (hi ? 8 : 0));
        a[i].u[1] = *(const uvec4*)(s + (hi ? 24 : 16));
      }
#pragma unroll
      for (int j = 0; j < 2; j++) {
        const unsigned short* s =
            &Bs[cur][(wn + j * 16 + lrow) * BK + ks + (hi ? 8 : 0)];
        bfr[j].u[0] = *(const uvec4*)(s);
        bfr[j].u[1] = *(const uvec4*)(s + 16);
      }
#pragma unroll
      for (int i = 0; i < 2; i++)
#pragma unroll
        for (int j = 0; j < 2; j++)
          acc[i][j] = __builtin_amdgcn_wmma_f32_16x16x32_bf16(
              false, a[i].v, false, bfr[j].v, (short)0, acc[i][j], false, false);
    }
    if (more) {
      if (GUARD) commit_guarded(cur ^ 1);
      else       wait_async0();
    }
    __syncthreads();
  }

  // -------- epilogue -------------------------------------------------------
#pragma unroll
  for (int i = 0; i < 2; i++)
#pragma unroll
    for (int j = 0; j < 2; j++)
#pragma unroll
      for (int r = 0; r < 8; r++) {
        int gm = m0 + wm + i * 16 + r + (hi ? 8 : 0);
        int gn = n0 + wn + j * 16 + lrow;
        if (gm < M && gn < N) {
          float v = acc[i][j][r] * alpha;
          if (bias)  v += bias[gn];
          if (resid) v += resid[(size_t)gm * ldr + gn];
          if (relu)  v = fmaxf(v, 0.0f);
          if (OUTTRANS) {
            if (OUTBF16) Cb[(size_t)gn * ldc + gm] = f2bf(v);
            else         Cf[(size_t)gn * ldc + gm] = v;
          } else {
            if (OUTBF16) Cb[(size_t)gm * ldc + gn] = f2bf(v);
            else         Cf[(size_t)gm * ldc + gn] = v;
          }
        }
      }
}

// ---------------------------------------------------------------------------
// Elementwise helpers
// ---------------------------------------------------------------------------
__global__ __launch_bounds__(256) void f32_to_bf16_kernel(
    const float* __restrict__ in, unsigned short* __restrict__ out, long long n) {
  long long i = (long long)blockIdx.x * 256 + threadIdx.x;
  if (i < n) out[i] = f2bf(in[i]);
}

// fp32 [K][N] -> bf16 [N][K] (transposed weight conversion)
__global__ __launch_bounds__(256) void f32_to_bf16_T_kernel(
    const float* __restrict__ in, unsigned short* __restrict__ out,
    int K, int N) {
  long long idx = (long long)blockIdx.x * 256 + threadIdx.x;
  if (idx >= (long long)K * N) return;
  int k = (int)(idx / N), n = (int)(idx - (long long)k * N);
  out[(size_t)n * K + k] = f2bf(in[idx]);
}

// kb = k * SCALE + rel_pos_bias broadcast (bias flat length 512 == H*DK)
__global__ __launch_bounds__(256) void kb_kernel(
    const unsigned short* __restrict__ k, const float* __restrict__ rpb,
    unsigned short* __restrict__ kb, long long n, float scale) {
  long long i = (long long)blockIdx.x * 256 + threadIdx.x;
  if (i < n) kb[i] = f2bf(bf2f(k[i]) * scale + rpb[i & 511]);
}

// LayerNorm: one 256-thread block per row, bf16 output.
__global__ __launch_bounds__(256) void layernorm_bf16(
    const float* __restrict__ in, const float* __restrict__ g,
    const float* __restrict__ bta, unsigned short* __restrict__ out, int cols) {
  __shared__ float red[256];
  const int row = blockIdx.x, t = threadIdx.x;
  const float* r = in + (size_t)row * cols;
  float s = 0.f;
  for (int c = t; c < cols; c += 256) s += r[c];
  red[t] = s; __syncthreads();
  for (int off = 128; off > 0; off >>= 1) {
    if (t < off) red[t] += red[t + off];
    __syncthreads();
  }
  float mean = red[0] / cols;
  __syncthreads();
  float v = 0.f;
  for (int c = t; c < cols; c += 256) { float d = r[c] - mean; v += d * d; }
  red[t] = v; __syncthreads();
  for (int off = 128; off > 0; off >>= 1) {
    if (t < off) red[t] += red[t + off];
    __syncthreads();
  }
  float rstd = rsqrtf(red[0] / cols + 1e-5f);
  for (int c = t; c < cols; c += 256)
    out[(size_t)row * cols + c] = f2bf((r[c] - mean) * rstd * g[c] + bta[c]);
}

// Enformer positional features: pos[4095 x 96], one thread per position.
__global__ __launch_bounds__(256) void pos_embed_kernel(float* __restrict__ pos) {
  int m = blockIdx.x * 256 + threadIdx.x;
  if (m >= 4095) return;
  const float d  = (float)(m - 2047);
  const float ad = fabsf(d);
  float emb[48];
#pragma unroll
  for (int n = 0; n < 16; n++) {                 // half_life = 2^linspace(3,11,16)
    float hl = exp2f(3.0f + (8.0f / 15.0f) * n);
    emb[n] = __expf(-0.6931471805599453f / hl * ad);
  }
#pragma unroll
  for (int n = 0; n < 16; n++)                   // cw = 2^(n+1) - 1
    emb[16 + n] = (exp2f((float)(n + 1)) - 1.0f > ad) ? 1.0f : 0.0f;
  float g[16], pm = 0.f;
#pragma unroll
  for (int n = 0; n < 16; n++) {                 // gamma pdf, mean=128(n+1), sd=64
    float mean = 128.0f * (n + 1);
    float conc = (mean / 64.0f) * (mean / 64.0f);
    float rate = mean / 4096.0f;
    float logp = (conc - 1.0f) * logf(ad) - rate * ad -
                 (lgammaf(conc) - conc * logf(rate));
    g[n] = __expf(logp) + 1e-8f;
    pm = fmaxf(pm, g[n]);
  }
#pragma unroll
  for (int n = 0; n < 16; n++) emb[32 + n] = g[n] / pm;
  float sign = (d > 0.f) ? 1.f : ((d < 0.f) ? -1.f : 0.f);
  float* row = pos + (size_t)m * 96;
#pragma unroll
  for (int f = 0; f < 48; f++) {
    row[f]      = emb[f];
    row[48 + f] = sign * emb[f];
  }
}

// rel_q[h][m][dk] = (pos @ Wrel)[m, h*64+dk], bf16 out. K = 96.
__global__ __launch_bounds__(256) void relq_kernel(
    const float* __restrict__ pos, const float* __restrict__ Wrel,
    unsigned short* __restrict__ relq) {
  int idx = blockIdx.x * 256 + threadIdx.x;
  if (idx >= 4095 * 512) return;
  int m = idx >> 9, c = idx & 511;
  const float* p = pos + (size_t)m * 96;
  float s = 0.f;
#pragma unroll 8
  for (int t = 0; t < 96; t++) s += p[t] * Wrel[(size_t)t * 512 + c];
  int h = c >> 6, dk = c & 63;
  relq[((size_t)h * 4095 + m) * 64 + dk] = f2bf(s);
}

// Softmax over j with fused relative-shift gather:
//   logits[z,i,j] += R[z, j, i - j + 511];  attn + attn^T written as bf16.
__global__ __launch_bounds__(256) void softmax_rel_kernel(
    const float* __restrict__ logits, const unsigned short* __restrict__ R,
    unsigned short* __restrict__ attn, unsigned short* __restrict__ attnT) {
  const int z    = blockIdx.y;
  const int wid  = threadIdx.x >> 5;
  const int lane = threadIdx.x & 31;
  const int i    = blockIdx.x * 8 + wid;     // < 2048
  const float* lrow = logits + ((size_t)z * 2048 + i) * 512;
  const unsigned short* Rz = R + (size_t)z * 512 * 4095;

  float vals[16];
  float mx = -3.0e38f;
#pragma unroll
  for (int t = 0; t < 16; t++) {
    int j = t * 32 + lane;
    float v = lrow[j] + bf2f(Rz[(size_t)j * 4095 + (i - j + 511)]);
    vals[t] = v;
    mx = fmaxf(mx, v);
  }
#pragma unroll
  for (int o = 16; o > 0; o >>= 1) mx = fmaxf(mx, __shfl_xor(mx, o, 32));
  float sum = 0.f;
#pragma unroll
  for (int t = 0; t < 16; t++) { vals[t] = __expf(vals[t] - mx); sum += vals[t]; }
#pragma unroll
  for (int o = 16; o > 0; o >>= 1) sum += __shfl_xor(sum, o, 32);
  float inv = 1.0f / sum;
#pragma unroll
  for (int t = 0; t < 16; t++) {
    int j = t * 32 + lane;
    unsigned short p = f2bf(vals[t] * inv);
    attn[((size_t)z * 2048 + i) * 512 + j]  = p;
    attnT[((size_t)z * 512 + j) * 2048 + i] = p;
  }
}

// ---------------------------------------------------------------------------
// Host side
// ---------------------------------------------------------------------------
static inline int cdiv(long long a, long long b) { return (int)((a + b - 1) / b); }

static void run_gemm(hipStream_t st, bool guard, bool outbf16, bool outtrans,
                     int M, int N, int K,
                     const unsigned short* A, int lda, long long sAb, long long sAh,
                     const unsigned short* B, int ldb, long long sBb, long long sBh,
                     void* C, int ldc, long long sCb, long long sCh,
                     const float* bias, const float* resid, int ldr,
                     float alpha, int relu, int batch, int Hnum) {
  dim3 grid(cdiv(N, BN), cdiv(M, BM), batch), blk(256);
  if (!guard && !outbf16 && !outtrans)
    gemm_tn_wmma<false, false, false><<<grid, blk, 0, st>>>(M, N, K, A, lda, sAb,
        sAh, B, ldb, sBb, sBh, C, ldc, sCb, sCh, bias, resid, ldr, alpha, relu, Hnum);
  else if (!guard && outbf16 && !outtrans)
    gemm_tn_wmma<false, true, false><<<grid, blk, 0, st>>>(M, N, K, A, lda, sAb,
        sAh, B, ldb, sBb, sBh, C, ldc, sCb, sCh, bias, resid, ldr, alpha, relu, Hnum);
  else if (!guard && outbf16 && outtrans)
    gemm_tn_wmma<false, true, true><<<grid, blk, 0, st>>>(M, N, K, A, lda, sAb,
        sAh, B, ldb, sBb, sBh, C, ldc, sCb, sCh, bias, resid, ldr, alpha, relu, Hnum);
  else if (guard && outbf16 && !outtrans)
    gemm_tn_wmma<true, true, false><<<grid, blk, 0, st>>>(M, N, K, A, lda, sAb,
        sAh, B, ldb, sBb, sBh, C, ldc, sCb, sCh, bias, resid, ldr, alpha, relu, Hnum);
  else
    gemm_tn_wmma<true, false, false><<<grid, blk, 0, st>>>(M, N, K, A, lda, sAb,
        sAh, B, ldb, sBb, sBh, C, ldc, sCb, sCh, bias, resid, ldr, alpha, relu, Hnum);
}

extern "C" void kernel_launch(void* const* d_in, const int* in_sizes, int n_in,
                              void* d_out, int out_size, void* d_ws, size_t ws_size,
                              hipStream_t stream) {
  (void)in_sizes; (void)n_in; (void)out_size; (void)ws_size;
  const int cB = 4, cN1 = 2048, cN2 = 512, cD = 768, cH = 8, cDK = 64;
  const int cDY0 = 1536, cLrel = 2 * cN1 - 1;  // 4095
  const float SCALE = 0.125f;                  // 64^-0.5

  const float* x     = (const float*)d_in[0];
  const float* y0    = (const float*)d_in[1];
  const float* W_res = (const float*)d_in[2];
  const float* lnx_g = (const float*)d_in[3];
  const float* lnx_b = (const float*)d_in[4];
  const float* lny_g = (const float*)d_in[5];
  const float* lny_b = (const float*)d_in[6];
  const float* Wq    = (const float*)d_in[7];
  const float* Wk    = (const float*)d_in[8];
  const float* Wv1   = (const float*)d_in[9];
  const float* Wv2   = (const float*)d_in[10];
  const float* Wo1   = (const float*)d_in[11];
  const float* bo1   = (const float*)d_in[12];
  const float* Wo2   = (const float*)d_in[13];
  const float* bo2   = (const float*)d_in[14];
  const float* Wrel  = (const float*)d_in[15];
  const float* rpb   = (const float*)d_in[16];
  const float* fx_g  = (const float*)d_in[17];
  const float* fx_b  = (const float*)d_in[18];
  const float* fx_w1 = (const float*)d_in[19];
  const float* fx_b1 = (const float*)d_in[20];
  const float* fx_w2 = (const float*)d_in[21];
  const float* fx_b2 = (const float*)d_in[22];
  const float* fy_g  = (const float*)d_in[23];
  const float* fy_b  = (const float*)d_in[24];
  const float* fy_w1 = (const float*)d_in[25];
  const float* fy_b1 = (const float*)d_in[26];
  const float* fy_w2 = (const float*)d_in[27];
  const float* fy_b2 = (const float*)d_in[28];

  // ---- workspace layout --------------------------------------------------
  char* ws = (char*)d_ws;
  size_t off = 0;
  auto alloc = [&](size_t bytes) -> char* {
    size_t a = (off + 255) & ~(size_t)255;
    off = a + bytes;
    return ws + a;
  };
  typedef unsigned short u16;
  const long long Mx = (long long)cB * cN1;        // 8192
  const long long My = (long long)cB * cN2;        // 2048
  u16* y0b   = (u16*)alloc((size_t)My * cDY0 * 2);
  u16* WresT = (u16*)alloc((size_t)cDY0 * cD * 2);     // [768][1536]
  u16* WqT   = (u16*)alloc((size_t)cD * 512 * 2);      // [512][768]
  u16* WkT   = (u16*)alloc((size_t)cD * 512 * 2);
  u16* Wv1T  = (u16*)alloc((size_t)cD * cD * 2);
  u16* Wv2T  = (u16*)alloc((size_t)cD * cD * 2);
  u16* Wo1T  = (u16*)alloc((size_t)cD * cD * 2);
  u16* Wo2T  = (u16*)alloc((size_t)cD * cD * 2);
  u16* fxw1T = (u16*)alloc((size_t)cD * 1536 * 2);     // [1536][768]
  u16* fxw2T = (u16*)alloc((size_t)1536 * cD * 2);     // [768][1536]
  u16* fyw1T = (u16*)alloc((size_t)cD * 1536 * 2);
  u16* fyw2T = (u16*)alloc((size_t)1536 * cD * 2);
  float* yf  = (float*)alloc((size_t)My * cD * 4);
  u16* x1b   = (u16*)alloc((size_t)Mx * cD * 2);
  u16* y1b   = (u16*)alloc((size_t)My * cD * 2);
  u16* qfb   = (u16*)alloc((size_t)Mx * 512 * 2);
  u16* kbuf  = (u16*)alloc((size_t)My * 512 * 2);
  u16* kbb   = (u16*)alloc((size_t)My * 512 * 2);
  u16* v2t   = (u16*)alloc((size_t)cB * cD * cN2 * 2); // [b][768][512]
  u16* v1t   = (u16*)alloc((size_t)cB * cD * cN1 * 2); // [b][768][2048]
  float* posf = (float*)alloc((size_t)cLrel * 96 * 4);
  u16* relqb = (u16*)alloc((size_t)cH * cLrel * cDK * 2);
  u16* Rb    = (u16*)alloc((size_t)cB * cH * cN2 * cLrel * 2);
  float* logitsf = (float*)alloc((size_t)cB * cH * cN1 * cN2 * 4);
  u16* attnb  = (u16*)alloc((size_t)cB * cH * cN1 * cN2 * 2);
  u16* attnTb = (u16*)alloc((size_t)cB * cH * cN2 * cN1 * 2);
  u16* out1b  = (u16*)alloc((size_t)Mx * cD * 2);
  u16* out2b  = (u16*)alloc((size_t)My * cD * 2);
  float* x4f  = (float*)alloc((size_t)Mx * cD * 4);
  float* y4f  = (float*)alloc((size_t)My * cD * 4);
  u16* x4nb   = (u16*)alloc((size_t)Mx * cD * 2);
  u16* y4nb   = (u16*)alloc((size_t)My * cD * 2);
  u16* hxb    = (u16*)alloc((size_t)Mx * 1536 * 2);
  u16* hyb    = (u16*)alloc((size_t)My * 1536 * 2);

  float* x5 = (float*)d_out;
  float* y5 = (float*)d_out + (size_t)Mx * cD;

  auto cvt = [&](const float* src, u16* dst, long long n) {
    f32_to_bf16_kernel<<<cdiv(n, 256), 256, 0, stream>>>(src, dst, n);
  };
  auto cvtT = [&](const float* src, u16* dst, int K, int N) {
    f32_to_bf16_T_kernel<<<cdiv((long long)K * N, 256), 256, 0, stream>>>(src,
                                                                          dst, K, N);
  };

  // ---- 0) conversions (weights transposed -> [N][K]) ---------------------
  cvt(y0, y0b, My * cDY0);
  cvtT(W_res, WresT, cDY0, cD);
  cvtT(Wq, WqT, cD, 512);
  cvtT(Wk, WkT, cD, 512);
  cvtT(Wv1, Wv1T, cD, cD);
  cvtT(Wv2, Wv2T, cD, cD);
  cvtT(Wo1, Wo1T, cD, cD);
  cvtT(Wo2, Wo2T, cD, cD);
  cvtT(fx_w1, fxw1T, cD, 1536);
  cvtT(fx_w2, fxw2T, 1536, cD);
  cvtT(fy_w1, fyw1T, cD, 1536);
  cvtT(fy_w2, fyw2T, 1536, cD);

  // ---- 1) y = y0 @ W_res -------------------------------------------------
  run_gemm(stream, false, false, false, (int)My, cD, cDY0,
           y0b, cDY0, 0, 0, WresT, cDY0, 0, 0, yf, cD, 0, 0,
           nullptr, nullptr, 0, 1.0f, 0, 1, 1);

  // ---- 2) LayerNorms -----------------------------------------------------
  layernorm_bf16<<<(int)Mx, 256, 0, stream>>>(x,  lnx_g, lnx_b, x1b, cD);
  layernorm_bf16<<<(int)My, 256, 0, stream>>>(yf, lny_g, lny_b, y1b, cD);

  // ---- 3) projections ----------------------------------------------------
  run_gemm(stream, false, true, false, (int)Mx, 512, cD, x1b, cD, 0, 0,
           WqT, cD, 0, 0, qfb, 512, 0, 0, nullptr, nullptr, 0, SCALE, 0, 1, 1);
  run_gemm(stream, false, true, false, (int)My, 512, cD, y1b, cD, 0, 0,
           WkT, cD, 0, 0, kbuf, 512, 0, 0, nullptr, nullptr, 0, 1.0f, 0, 1, 1);
  // v2^T per batch: [768][512];  v1^T per batch: [768][2048]
  run_gemm(stream, false, true, true, cN2, cD, cD,
           y1b, cD, (long long)cN2 * cD, 0,
           Wv2T, cD, 0, 0,
           v2t, cN2, (long long)cD * cN2, 0,
           nullptr, nullptr, 0, 1.0f, 0, cB, 1);
  run_gemm(stream, false, true, true, cN1, cD, cD,
           x1b, cD, (long long)cN1 * cD, 0,
           Wv1T, cD, 0, 0,
           v1t, cN1, (long long)cD * cN1, 0,
           nullptr, nullptr, 0, 1.0f, 0, cB, 1);

  kb_kernel<<<cdiv(My * 512, 256), 256, 0, stream>>>(kbuf, rpb, kbb,
                                                     My * 512, SCALE);

  // ---- 4) positional features & rel_q ------------------------------------
  pos_embed_kernel<<<cdiv(cLrel, 256), 256, 0, stream>>>(posf);
  relq_kernel<<<cdiv((long long)cLrel * 512, 256), 256, 0, stream>>>(posf, Wrel,
                                                                     relqb);

  // ---- 5) R[z] = kb[z] @ rel_q[h]^T   ([512 x 4095], bf16, guarded) ------
  run_gemm(stream, true, true, false, cN2, cLrel, cDK,
           kbb, 512, (long long)cN2 * 512, 64,
           relqb, 64, 0, (long long)cLrel * 64,
           Rb, cLrel, (long long)cH * cN2 * cLrel, (long long)cN2 * cLrel,
           nullptr, nullptr, 0, 1.0f, 0, cB * cH, cH);

  // ---- 6) content logits[z] = q[z] @ k[z]^T (fp32, async path) -----------
  run_gemm(stream, false, false, false, cN1, cN2, cDK,
           qfb, 512, (long long)cN1 * 512, 64,
           kbuf, 512, (long long)cN2 * 512, 64,
           logitsf, cN2, (long long)cH * cN1 * cN2, (long long)cN1 * cN2,
           nullptr, nullptr, 0, 1.0f, 0, cB * cH, cH);

  // ---- 7) softmax with fused relative-shift gather -----------------------
  {
    dim3 grid(cN1 / 8, cB * cH);
    softmax_rel_kernel<<<grid, 256, 0, stream>>>(logitsf, Rb, attnb, attnTb);
  }

  // ---- 8) out1 = attn @ v2 ; out2 = attn^T @ v1 (guarded, N=96) ----------
  run_gemm(stream, true, true, false, cN1, 96, cN2,
           attnb, 512, (long long)cH * cN1 * cN2, (long long)cN1 * cN2,
           v2t, cN2, (long long)cD * cN2, (long long)96 * cN2,
           out1b, cD, (long long)cN1 * cD, 96,
           nullptr, nullptr, 0, 1.0f, 0, cB * cH, cH);
  run_gemm(stream, true, true, false, cN2, 96, cN1,
           attnTb, cN1, (long long)cH * cN2 * cN1, (long long)cN2 * cN1,
           v1t, cN1, (long long)cD * cN1, (long long)96 * cN1,
           out2b, cD, (long long)cN2 * cD, 96,
           nullptr, nullptr, 0, 1.0f, 0, cB * cH, cH);

  // ---- 9) output projections + residual ----------------------------------
  run_gemm(stream, false, false, false, (int)Mx, cD, cD, out1b, cD, 0, 0,
           Wo1T, cD, 0, 0, x4f, cD, 0, 0, bo1, x, cD, 1.0f, 0, 1, 1);
  run_gemm(stream, false, false, false, (int)My, cD, cD, out2b, cD, 0, 0,
           Wo2T, cD, 0, 0, y4f, cD, 0, 0, bo2, yf, cD, 1.0f, 0, 1, 1);

  // ---- 10) FFNs -----------------------------------------------------------
  layernorm_bf16<<<(int)Mx, 256, 0, stream>>>(x4f, fx_g, fx_b, x4nb, cD);
  layernorm_bf16<<<(int)My, 256, 0, stream>>>(y4f, fy_g, fy_b, y4nb, cD);

  run_gemm(stream, false, true, false, (int)Mx, 1536, cD, x4nb, cD, 0, 0,
           fxw1T, cD, 0, 0, hxb, 1536, 0, 0, fx_b1, nullptr, 0, 1.0f, 1, 1, 1);
  run_gemm(stream, false, false, false, (int)Mx, cD, 1536, hxb, 1536, 0, 0,
           fxw2T, 1536, 0, 0, x5, cD, 0, 0, fx_b2, x4f, cD, 1.0f, 0, 1, 1);

  run_gemm(stream, false, true, false, (int)My, 1536, cD, y4nb, cD, 0, 0,
           fyw1T, cD, 0, 0, hyb, 1536, 0, 0, fy_b1, nullptr, 0, 1.0f, 1, 1, 1);
  run_gemm(stream, false, false, false, (int)My, cD, 1536, hyb, 1536, 0, 0,
           fyw2T, 1536, 0, 0, y5, cD, 0, 0, fy_b2, y4f, cD, 1.0f, 0, 1, 1);
}
